// MoEDictionaryCrossAttention_16518444220937
// MI455X (gfx1250) — matrix operand order repaired and checked by reading, not compiled, not run on hardware
//
#include <hip/hip_runtime.h>
#include <hip/hip_bf16.h>

// ---------------------------------------------------------------------------
// MoE Dictionary Cross-Attention forward for MI455X (gfx1250, wave32, WMMA).
// All large matmuls run on v_wmma_f32_16x16x32_f16 with f16 activations/weights
// and fused epilogues; softmax attention is a fused WMMA kernel.
// GEMM staging uses GLOBAL_LOAD_ASYNC_TO_LDS_B128 (ASYNCcnt) with LDS double
// buffering when the toolchain exposes the builtin.
// ---------------------------------------------------------------------------

typedef _Float16 f16;
typedef _Float16 v16h __attribute__((ext_vector_type(16)));
typedef _Float16 v8h  __attribute__((ext_vector_type(8)));
typedef float    v8f  __attribute__((ext_vector_type(8)));
typedef int      i32x4 __attribute__((vector_size(16)));
typedef __attribute__((address_space(1))) i32x4 gas_i32x4;
typedef __attribute__((address_space(3))) i32x4 las_i32x4;

#define Bsz   4
#define Hd    64
#define Wd    64
#define Ssz   16384          // B*H*W
#define INC   320
#define OUTC  320
#define Dd    640
#define Ed    20
#define Cd    32
#define KEXP  4
#define ENTd  64
#define HIDd  1280
#define CATd  2560
#define R1d   160

#if defined(__HIP_DEVICE_COMPILE__) && defined(__gfx1250__)
#  if __has_builtin(__builtin_amdgcn_global_load_async_to_lds_b128)
#    define HAS_ASYNC_LDS 1
#  else
#    define HAS_ASYNC_LDS 0
#  endif
#  if __has_builtin(__builtin_amdgcn_s_wait_asynccnt)
#    define HAS_WAIT_ASYNC 1
#  else
#    define HAS_WAIT_ASYNC 0
#  endif
#else
#  define HAS_ASYNC_LDS 0
#  define HAS_WAIT_ASYNC 0
#endif

__device__ __forceinline__ v8f zero8() {
    v8f z = {0.f,0.f,0.f,0.f,0.f,0.f,0.f,0.f};
    return z;
}
__device__ __forceinline__ v16h cat8(v8h a, v8h b) {
    return __builtin_shufflevector(a, b, 0,1,2,3,4,5,6,7,8,9,10,11,12,13,14,15);
}
__device__ __forceinline__ v8f wmma_f16(v16h a, v16h b, v8f c) {
    // D = A(16x32) * B(32x16) + C  -> v_wmma_f32_16x16x32_f16
    return __builtin_amdgcn_wmma_f32_16x16x32_f16(false, a, false, b, (short)0, c, false, false);
}
__device__ __forceinline__ float gelu_f(float x) {
    return 0.5f * x * (1.f + erff(x * 0.70710678118654752f));
}

// Copy 32 contiguous bytes global -> LDS. Async path: two b128 async-to-LDS
// ops sharing base pointers, second uses inst_offset 16 (applied to both
// addresses per the ISA). Fallback: VGPR round-trip.
__device__ __forceinline__ void cp32_g2l(const f16* __restrict__ g, f16* l) {
#if HAS_ASYNC_LDS
    __builtin_amdgcn_global_load_async_to_lds_b128(
        (gas_i32x4*)g, (las_i32x4*)l, 0, 0);
    __builtin_amdgcn_global_load_async_to_lds_b128(
        (gas_i32x4*)g, (las_i32x4*)l, 16, 0);
#else
    const uint4* s = (const uint4*)g;
    uint4 a = s[0], b = s[1];
    uint4* d = (uint4*)l;
    d[0] = a; d[1] = b;
#endif
}
__device__ __forceinline__ void async_wait_all() {
#if HAS_ASYNC_LDS
#  if HAS_WAIT_ASYNC
    __builtin_amdgcn_s_wait_asynccnt(0);
#  else
    asm volatile("s_wait_asynccnt 0x0" ::: "memory");
#  endif
#endif
}

// ---------------------------------------------------------------------------
// Generic WMMA GEMM:  Y[M,N] = A[M,K] @ Bt[N,K]^T  (+bias, act, residual)
// A: f16 row-major (lda), Bt: f16 [N][K] row-major (ldb == K).
// Block: 128x128 tile, 256 threads = 8 waves (4 M-waves x 2 N-waves),
// each wave computes 2x4 tiles of 16x16, K stepped by 32.
// LDS double-buffered; next K-tile staged asynchronously during compute.
// ---------------------------------------------------------------------------
template<int ACT, bool RES, bool NCHW>
__global__ __launch_bounds__(256) void wmma_gemm(
    const f16* __restrict__ A, const f16* __restrict__ Bt,
    int M, int N, int Kd2, long lda, long ldb,
    const float* __restrict__ bias,
    float* __restrict__ out32, f16* __restrict__ out16, long ldc,
    const float* __restrict__ res, const float* __restrict__ rsv, long ldres,
    int nHW)
{
    __shared__ alignas(32) f16 sA[2][128 * 32];
    __shared__ alignas(32) f16 sB[2][128 * 32];

    const int bm = blockIdx.y * 128, bn = blockIdx.x * 128;
    const int tid  = threadIdx.x;
    const int wave = tid >> 5, lane = tid & 31;
    const int wm = wave & 3, wn = wave >> 2;
    const int half = lane >> 4, l16 = lane & 15;

    v8f acc[2][4];
    #pragma unroll
    for (int i = 0; i < 2; ++i)
        #pragma unroll
        for (int j = 0; j < 4; ++j) acc[i][j] = zero8();

    // staging assignment: thread -> (row = tid/2, 16 f16 at col (tid&1)*16)
    const int ar  = tid >> 1;
    const int acs = (tid & 1) * 16;
    int garow = bm + ar; if (garow >= M) garow = M - 1;
    int gbrow = bn + ar; if (gbrow >= N) gbrow = N - 1;
    const f16* gA = A  + (long)garow * lda + acs;
    const f16* gB = Bt + (long)gbrow * ldb + acs;

    auto stage = [&](int k0, int buf) {
        cp32_g2l(gA + k0, &sA[buf][ar * 32 + acs]);
        cp32_g2l(gB + k0, &sB[buf][ar * 32 + acs]);
    };

    stage(0, 0);
    async_wait_all();
    __syncthreads();

    const int nk = Kd2 >> 5;
    for (int kk = 0; kk < nk; ++kk) {
        const int buf = kk & 1;
        if (kk + 1 < nk) stage((kk + 1) << 5, buf ^ 1);   // overlap with compute

        const f16* cA = &sA[buf][0];
        const f16* cB = &sB[buf][0];
        v16h af[2], bf[4];
        #pragma unroll
        for (int tm = 0; tm < 2; ++tm) {
            int rowA = wm * 32 + tm * 16 + l16;
            v8h a0 = *(const v8h*)&cA[rowA * 32 + 16 * half];
            v8h a1 = *(const v8h*)&cA[rowA * 32 + 8 + 16 * half];
            af[tm] = cat8(a0, a1);
        }
        #pragma unroll
        for (int tn = 0; tn < 4; ++tn) {
            int rowB = wn * 64 + tn * 16 + l16;
            bf[tn] = *(const v16h*)&cB[rowB * 32 + 16 * half];
        }
        #pragma unroll
        for (int tm = 0; tm < 2; ++tm)
            #pragma unroll
            for (int tn = 0; tn < 4; ++tn)
                acc[tm][tn] = wmma_f16(af[tm], bf[tn], acc[tm][tn]);

        async_wait_all();   // next tile resident
        __syncthreads();    // all waves done with buf, next buf visible
    }

    // epilogue: C layout -> element (r, lane): M = r + 8*half, N = lane%16
    #pragma unroll
    for (int tm = 0; tm < 2; ++tm) {
        #pragma unroll
        for (int tn = 0; tn < 4; ++tn) {
            int gcol = bn + wn * 64 + tn * 16 + l16;
            if (gcol >= N) continue;
            float bia = bias ? bias[gcol] : 0.f;
            float rw  = RES ? rsv[gcol] : 0.f;
            #pragma unroll
            for (int r = 0; r < 8; ++r) {
                int grow = bm + wm * 32 + tm * 16 + 8 * half + r;
                if (grow >= M) continue;
                float v = acc[tm][tn][r] + bia;
                if (ACT == 1) v = fmaxf(v, 0.f);
                if (ACT == 2) v = gelu_f(v);
                if (RES) v += res[(long)grow * ldres + gcol] * rw;
                if (NCHW) {
                    int bb = grow / nHW, hw = grow % nHW;
                    out32[((long)bb * N + gcol) * (long)nHW + hw] = v;
                } else {
                    if (out32) out32[(long)grow * ldc + gcol] = v;
                    if (out16) out16[(long)grow * ldc + gcol] = (f16)v;
                }
            }
        }
    }
}

// ---------------------------------------------------------------------------
// Fused soft-MoE dictionary cross-attention (WMMA).
// Grid: (S/128 per batch, B*E). Each wave owns 16 rows; loops 4 experts,
// accumulating gate-weighted eo into ctx.
// ---------------------------------------------------------------------------
__global__ __launch_bounds__(256) void k_attention(
    const f16* __restrict__ qh, const f16* __restrict__ kxh,
    const f16* __restrict__ vxT, const float* __restrict__ gates,
    const float* __restrict__ scale_attn, f16* __restrict__ ctx)
{
    __shared__ alignas(32) f16 sQ[128 * 32];
    __shared__ alignas(32) f16 sK[64 * 32];
    __shared__ alignas(32) f16 sV[32 * 64];
    __shared__ alignas(32) f16 sP[8][16 * 64];

    const int be = blockIdx.y;
    const int e = be % Ed, b = be / Ed;
    const int s0 = blockIdx.x * 128;
    const int tid = threadIdx.x, wave = tid >> 5, lane = tid & 31;
    const int half = lane >> 4, l16 = lane & 15;

    {   // stage Q tile [128][32] from qh[S][640] columns e*32..+32
        int r = tid >> 1, cs = (tid & 1) * 16;
        cp32_g2l(qh + ((long)(b * 4096 + s0 + r)) * Dd + e * Cd + cs, &sQ[r * 32 + cs]);
    }
    const float sc = scale_attn[e];
    v8f cacc[2]; cacc[0] = zero8(); cacc[1] = zero8();

    for (int n = 0; n < KEXP; ++n) {
        __syncthreads();
        if (tid < 128) {            // stage kx[n][e]: [64][32] f16 (Bt layout)
            int r = tid >> 1, cs = (tid & 1) * 16;
            cp32_g2l(kxh + ((long)((n * Ed + e) * ENTd + r)) * Cd + cs, &sK[r * 32 + cs]);
        } else {                    // stage vxT[n][e]: [32][64] f16 (Bt layout)
            int t2 = tid - 128;
            int r = t2 >> 2, cs = (t2 & 3) * 16;
            cp32_g2l(vxT + ((long)((n * Ed + e) * Cd + r)) * ENTd + cs, &sV[r * 64 + cs]);
        }
        async_wait_all();
        __syncthreads();

        // sim = q(16x32) @ kx^T(32x64): 4 WMMAs
        int rq = wave * 16 + l16;
        v16h aq;
        {
            v8h a0 = *(const v8h*)&sQ[rq * 32 + 16 * half];
            v8h a1 = *(const v8h*)&sQ[rq * 32 + 8 + 16 * half];
            aq = cat8(a0, a1);
        }
        v8f sim[4];
        #pragma unroll
        for (int t = 0; t < 4; ++t) {
            v16h bk = *(const v16h*)&sK[(t * 16 + l16) * 32 + 16 * half];
            sim[t] = wmma_f16(aq, bk, zero8());
        }

        // softmax over ENT=64 per row (reduce 16 lanes x 4 accumulators)
        float pr[4][8];
        #pragma unroll
        for (int r = 0; r < 8; ++r) {
            float m = -1e30f;
            #pragma unroll
            for (int t = 0; t < 4; ++t) { float v = sim[t][r] * sc; pr[t][r] = v; m = fmaxf(m, v); }
            #pragma unroll
            for (int msk = 1; msk < 16; msk <<= 1) m = fmaxf(m, __shfl_xor(m, msk, 32));
            float su = 0.f;
            #pragma unroll
            for (int t = 0; t < 4; ++t) { float v = expf(pr[t][r] - m); pr[t][r] = v; su += v; }
            #pragma unroll
            for (int msk = 1; msk < 16; msk <<= 1) su += __shfl_xor(su, msk, 32);
            float inv = 1.f / su;
            #pragma unroll
            for (int t = 0; t < 4; ++t) pr[t][r] *= inv;
        }

        // reshape probs via wave-private LDS into A-fragment layout
        f16* P = &sP[wave][0];
        #pragma unroll
        for (int t = 0; t < 4; ++t)
            #pragma unroll
            for (int r = 0; r < 8; ++r)
                P[(r + 8 * half) * 64 + t * 16 + l16] = (f16)pr[t][r];

        // eo = probs(16x64) @ vx(64x32): 2 K-steps x 2 N-tiles
        v8f eacc[2]; eacc[0] = zero8(); eacc[1] = zero8();
        #pragma unroll
        for (int kk = 0; kk < 2; ++kk) {
            v16h ap;
            {
                v8h p0 = *(const v8h*)&P[l16 * 64 + kk * 32 + 16 * half];
                v8h p1 = *(const v8h*)&P[l16 * 64 + kk * 32 + 8 + 16 * half];
                ap = cat8(p0, p1);
            }
            #pragma unroll
            for (int tn = 0; tn < 2; ++tn) {
                v16h bv = *(const v16h*)&sV[(tn * 16 + l16) * 64 + kk * 32 + 16 * half];
                eacc[tn] = wmma_f16(ap, bv, eacc[tn]);
            }
        }

        // gate-weighted accumulate
        long srow = (long)b * 4096 + s0 + wave * 16;
        float g[8];
        #pragma unroll
        for (int r = 0; r < 8; ++r) g[r] = gates[(srow + r + 8 * half) * KEXP + n];
        #pragma unroll
        for (int tn = 0; tn < 2; ++tn)
            #pragma unroll
            for (int r = 0; r < 8; ++r) cacc[tn][r] += g[r] * eacc[tn][r];
    }

    #pragma unroll
    for (int tn = 0; tn < 2; ++tn)
        #pragma unroll
        for (int r = 0; r < 8; ++r) {
            long row = (long)b * 4096 + s0 + wave * 16 + r + 8 * half;
            ctx[row * Dd + e * Cd + tn * 16 + l16] = (f16)cacc[tn][r];
        }
}

// ---------------------------------------------------------------------------
// Helper kernels (elementwise / small ops)
// ---------------------------------------------------------------------------
__global__ void k_wtrans(const float* __restrict__ w, f16* __restrict__ wt, int Kd2, int N) {
    long i = (long)blockIdx.x * 256 + threadIdx.x;
    if (i >= (long)N * Kd2) return;
    int n = (int)(i / Kd2), k = (int)(i % Kd2);
    wt[i] = (f16)w[(long)k * N + n];
}
__global__ void k_wcast(const float* __restrict__ w, f16* __restrict__ wt, long n) {
    long i = (long)blockIdx.x * 256 + threadIdx.x;
    if (i >= n) return;
    wt[i] = (f16)w[i];
}
__global__ void k_nchw2nhwc(const float* __restrict__ x, f16* __restrict__ xh) {
    long i = (long)blockIdx.x * 256 + threadIdx.x;
    if (i >= (long)Ssz * INC) return;
    int c = (int)(i % INC); long t = i / INC;
    int hw = (int)(t % 4096); int b = (int)(t / 4096);
    xh[i] = (f16)x[((long)b * INC + c) * 4096 + hw];
}
__global__ __launch_bounds__(256) void k_layernorm(
    const float* __restrict__ in, const float* __restrict__ g, const float* __restrict__ be,
    int rows, int dim, f16* __restrict__ o16, float* __restrict__ o32)
{
    int row = blockIdx.x * 8 + (threadIdx.x >> 5);
    int lane = threadIdx.x & 31;
    if (row >= rows) return;
    const float* x = in + (long)row * dim;
    float s = 0.f, ss = 0.f;
    for (int c = lane; c < dim; c += 32) { float v = x[c]; s += v; ss += v * v; }
    for (int m = 16; m >= 1; m >>= 1) { s += __shfl_xor(s, m, 32); ss += __shfl_xor(ss, m, 32); }
    float mean = s / dim, var = ss / dim - mean * mean;
    float inv = rsqrtf(var + 1e-5f);
    for (int c = lane; c < dim; c += 32) {
        float v = (x[c] - mean) * inv * g[c] + be[c];
        if (o16) o16[(long)row * dim + c] = (f16)v;
        if (o32) o32[(long)row * dim + c] = v;
    }
}
__global__ void k_gelu16(const f16* __restrict__ in, long ldi, f16* __restrict__ out,
                         long ldo, long rows, int cols) {
    long i = (long)blockIdx.x * 256 + threadIdx.x;
    if (i >= rows * cols) return;
    int c = (int)(i % cols); long r = i / cols;
    out[r * ldo + c] = (f16)gelu_f((float)in[r * ldi + c]);
}
// Depthwise 3x3, NHWC. MODE 0: out = gelu(dw(gelu(x)) + b) (MSA dense block)
//                     MODE 1: out = gelu(dw(x) + b) * x[:, vofs+c] (gated MLP)
template<int MODE>
__global__ void k_dw3x3(const float* __restrict__ in, long ldin,
                        const float* __restrict__ w, const float* __restrict__ bias,
                        f16* __restrict__ out, int Ch, int vofs)
{
    long i = (long)blockIdx.x * 256 + threadIdx.x;
    if (i >= (long)Ssz * Ch) return;
    int c = (int)(i % Ch); long s = i / Ch;
    int x0 = (int)(s & 63), y0 = (int)((s >> 6) & 63), b = (int)(s >> 12);
    float acc = bias[c];
    #pragma unroll
    for (int dy = -1; dy <= 1; ++dy) {
        int yy = y0 + dy; if ((unsigned)yy >= 64u) continue;
        #pragma unroll
        for (int dx = -1; dx <= 1; ++dx) {
            int xx = x0 + dx; if ((unsigned)xx >= 64u) continue;
            long ss = ((long)b * 64 + yy) * 64 + xx;
            float v = in[ss * ldin + c];
            if (MODE == 0) v = gelu_f(v);
            acc += v * w[c * 9 + (dy + 1) * 3 + (dx + 1)];
        }
    }
    float o = gelu_f(acc);
    if (MODE == 1) o *= in[s * ldin + vofs + c];
    out[s * (long)Ch + c] = (f16)o;
}
__global__ __launch_bounds__(256) void k_avgmax(const float* __restrict__ z, float* __restrict__ am) {
    int row = blockIdx.x * 8 + (threadIdx.x >> 5);
    int lane = threadIdx.x & 31;
    const float* x = z + (long)row * Dd;
    float s = 0.f, m = -1e30f;
    for (int c = lane; c < Dd; c += 32) { float v = x[c]; s += v; m = fmaxf(m, v); }
    for (int k = 16; k >= 1; k >>= 1) { s += __shfl_xor(s, k, 32); m = fmaxf(m, __shfl_xor(m, k, 32)); }
    if (lane == 0) { am[row * 2] = s / (float)Dd; am[row * 2 + 1] = m; }
}
__global__ void k_spatatt(const float* __restrict__ am, const float* __restrict__ w,
                          float* __restrict__ att) {
    int s = blockIdx.x * 256 + threadIdx.x;
    if (s >= Ssz) return;
    int x = s & 63, y = (s >> 6) & 63, b = s >> 12;
    float acc = 0.f;
    for (int ch = 0; ch < 2; ++ch)
        for (int dy = -3; dy <= 3; ++dy) {
            int yy = y + dy; if ((unsigned)yy >= 64u) continue;
            for (int dx = -3; dx <= 3; ++dx) {
                int xx = x + dx; if ((unsigned)xx >= 64u) continue;
                acc += am[(((long)b * 64 + yy) * 64 + xx) * 2 + ch] * w[ch * 49 + (dy + 3) * 7 + (dx + 3)];
            }
        }
    att[s] = 1.f / (1.f + expf(-acc));
}
__global__ void k_msacomb(const float* __restrict__ z, const float* __restrict__ att,
                          const float* __restrict__ x1, const float* __restrict__ rs1,
                          float* __restrict__ x2, f16* __restrict__ x2h) {
    long i = (long)blockIdx.x * 256 + threadIdx.x;
    if (i >= (long)Ssz * Dd) return;
    int c = (int)(i % Dd); long s = i / Dd;
    float v = z[i] * att[s] + x1[i] * rs1[c];
    x2[i] = v; x2h[i] = (f16)v;
}
__global__ void k_router2(const f16* __restrict__ r16, const float* __restrict__ w2,
                          const float* __restrict__ b2, float* __restrict__ gates) {
    int s = blockIdx.x * 256 + threadIdx.x;
    if (s >= Ssz) return;
    float l[4] = { b2[0], b2[1], b2[2], b2[3] };
    const f16* rp = r16 + (long)s * R1d;
    for (int i = 0; i < R1d; ++i) {
        float rv = (float)rp[i];
        #pragma unroll
        for (int j = 0; j < 4; ++j) l[j] += rv * w2[i * 4 + j];
    }
    float m = fmaxf(fmaxf(l[0], l[1]), fmaxf(l[2], l[3]));
    float su = 0.f;
    #pragma unroll
    for (int j = 0; j < 4; ++j) { l[j] = expf(l[j] - m); su += l[j]; }
    float inv = 1.f / su;
    #pragma unroll
    for (int j = 0; j < 4; ++j) gates[(long)s * 4 + j] = l[j] * inv;
}
// Expert K/V prep: kx = LN(experts)@k_w + k_b permuted to [n][e][ENT][C] f16;
// vxT = LN(experts) permuted/transposed to [n][e][C][ENT] f16.
__global__ void k_expert_kv(const float* __restrict__ ed, const float* __restrict__ kw,
                            const float* __restrict__ kb, f16* __restrict__ kxh,
                            f16* __restrict__ vxT) {
    long i = (long)blockIdx.x * 256 + threadIdx.x;
    if (i >= (long)KEXP * ENTd * Dd) return;
    int d = (int)(i % Dd); long t = i / Dd;
    int ent = (int)(t % ENTd); int n = (int)(t / ENTd);
    int e = d >> 5, c = d & 31;
    const float* row = ed + ((long)n * ENTd + ent) * Dd;
    float acc = kb[d];
    for (int j = 0; j < Dd; ++j) acc += row[j] * kw[(long)j * Dd + d];
    kxh[(((long)(n * Ed + e) * ENTd + ent)) * Cd + c] = (f16)acc;
    vxT[(((long)(n * Ed + e) * Cd + c)) * ENTd + ent] = (f16)row[d];
}

// ---------------------------------------------------------------------------
extern "C" void kernel_launch(void* const* d_in, const int* in_sizes, int n_in,
                              void* d_out, int out_size, void* d_ws, size_t ws_size,
                              hipStream_t stream)
{
    (void)in_sizes; (void)n_in; (void)out_size; (void)ws_size;
    auto F = [&](int i) { return (const float*)d_in[i]; };
    // d_in order: 0 x, then params in insertion order
    // 1 experts 2 router_w1 3 router_b1 4 router_w2 5 router_b2 6 scale_attn
    // 7 x_trans_w 8 x_trans_b 9 ln_scale_g 10 ln_scale_b 11 msa_s_w 12 msa_s_b
    // 13 dense_proj_w 14 dense_proj_b 15 sa_w 16 lnx_g 17 lnx_b 18 q_w 19 q_b
    // 20 dict_g 21 dict_b 22 k_w 23 k_b 24 lin_w 25 lin_b 26 ln_mlp_g 27 ln_mlp_b
    // 28 fc1_w 29 fc1_b 30 mlp_dw_w 31 mlp_dw_b 32 fc2_w 33 fc2_b 34 out_w 35 out_b
    // 36 rs1 37 rs2 38 rs3; 39+6i: dw{i}_in_w,in_b,dw_w,dw_b,out_w,out_b

    char* wp = (char*)d_ws;
    auto alc = [&](size_t bytes) { void* r = (void*)wp; wp += (bytes + 255) & ~(size_t)255; return r; };

    f16* wt_xt   = (f16*)alc((size_t)Dd * INC * 2);
    f16* wt_msas = (f16*)alc((size_t)Dd * Dd * 2);
    f16* wt_dp   = (f16*)alc((size_t)Dd * CATd * 2);
    f16* wt_r1   = (f16*)alc((size_t)R1d * Dd * 2);
    f16* wt_q    = (f16*)alc((size_t)Dd * Dd * 2);
    f16* wt_lin  = (f16*)alc((size_t)Dd * Dd * 2);
    f16* wt_fc1  = (f16*)alc((size_t)CATd * Dd * 2);
    f16* wt_fc2  = (f16*)alc((size_t)Dd * HIDd * 2);
    f16* wt_out  = (f16*)alc((size_t)OUTC * Dd * 2);
    f16 *wt_dwin[3], *wt_dwout[3];
    for (int i = 0; i < 3; ++i) {
        wt_dwin[i]  = (f16*)alc((size_t)Dd * Dd * 2);
        wt_dwout[i] = (f16*)alc((size_t)Dd * Dd * 2);
    }
    f16*   xh   = (f16*)alc((size_t)Ssz * INC * 2);
    float* x1   = (float*)alc((size_t)Ssz * Dd * 4);
    f16*   a16  = (f16*)alc((size_t)Ssz * Dd * 2);
    f16*   cat  = (f16*)alc((size_t)Ssz * CATd * 2);
    f16*   t16  = (f16*)alc((size_t)Ssz * Dd * 2);
    float* t1   = (float*)alc((size_t)Ssz * Dd * 4);
    f16*   t2h  = (f16*)alc((size_t)Ssz * Dd * 2);
    float* zb   = (float*)alc((size_t)Ssz * Dd * 4);
    float* am   = (float*)alc((size_t)Ssz * 2 * 4);
    float* att  = (float*)alc((size_t)Ssz * 4);
    float* x2   = (float*)alc((size_t)Ssz * Dd * 4);
    f16*   x2h  = (f16*)alc((size_t)Ssz * Dd * 2);
    f16*   r16  = (f16*)alc((size_t)Ssz * R1d * 2);
    float* gates= (float*)alc((size_t)Ssz * 4 * 4);
    f16*   qh   = (f16*)alc((size_t)Ssz * Dd * 2);
    float* ed   = (float*)alc((size_t)KEXP * ENTd * Dd * 4);
    f16*   kxh  = (f16*)alc((size_t)KEXP * Ed * ENTd * Cd * 2);
    f16*   vxT  = (f16*)alc((size_t)KEXP * Ed * Cd * ENTd * 2);
    f16*   ctxh = (f16*)alc((size_t)Ssz * Dd * 2);
    float* o2   = (float*)alc((size_t)Ssz * Dd * 4);
    float* hb   = (float*)alc((size_t)Ssz * CATd * 4);
    f16*   mmid = (f16*)alc((size_t)Ssz * HIDd * 2);
    float* o3   = (float*)alc((size_t)Ssz * Dd * 4);
    f16*   o3h  = (f16*)alc((size_t)Ssz * Dd * 2);

    dim3 blk(256);
    auto cdiv = [](long a, long b) { return (unsigned)((a + b - 1) / b); };

    auto gemm = [&](const f16* A, long lda, const f16* Bt, int M, int N, int Kd2,
                    const float* bias, float* o32, f16* o16, long ldc,
                    int act, const float* res, const float* rsv, bool nchw) {
        dim3 g(cdiv(N, 128), cdiv(M, 128));
        if (nchw)
            wmma_gemm<0,false,true ><<<g, blk, 0, stream>>>(A, Bt, M, N, Kd2, lda, Kd2, bias,
                o32, o16, ldc, nullptr, nullptr, 0, 4096);
        else if (res)
            wmma_gemm<0,true ,false><<<g, blk, 0, stream>>>(A, Bt, M, N, Kd2, lda, Kd2, bias,
                o32, o16, ldc, res, rsv, Dd, 0);
        else if (act == 1)
            wmma_gemm<1,false,false><<<g, blk, 0, stream>>>(A, Bt, M, N, Kd2, lda, Kd2, bias,
                o32, o16, ldc, nullptr, nullptr, 0, 0);
        else
            wmma_gemm<0,false,false><<<g, blk, 0, stream>>>(A, Bt, M, N, Kd2, lda, Kd2, bias,
                o32, o16, ldc, nullptr, nullptr, 0, 0);
    };

    // ---- weight prep: transpose linear weights to [N][K] f16; cast conv 1x1
    k_wtrans<<<cdiv((long)Dd * INC, 256), blk, 0, stream>>>(F(7),  wt_xt,  INC,  Dd);
    k_wtrans<<<cdiv((long)R1d * Dd, 256), blk, 0, stream>>>(F(2),  wt_r1,  Dd,   R1d);
    k_wtrans<<<cdiv((long)Dd * Dd, 256),  blk, 0, stream>>>(F(18), wt_q,   Dd,   Dd);
    k_wtrans<<<cdiv((long)Dd * Dd, 256),  blk, 0, stream>>>(F(24), wt_lin, Dd,   Dd);
    k_wtrans<<<cdiv((long)CATd * Dd, 256),blk, 0, stream>>>(F(28), wt_fc1, Dd,   CATd);
    k_wtrans<<<cdiv((long)Dd * HIDd, 256),blk, 0, stream>>>(F(32), wt_fc2, HIDd, Dd);
    k_wtrans<<<cdiv((long)OUTC * Dd, 256),blk, 0, stream>>>(F(34), wt_out, Dd,   OUTC);
    k_wcast<<<cdiv((long)Dd * Dd, 256),   blk, 0, stream>>>(F(11), wt_msas, (long)Dd * Dd);
    k_wcast<<<cdiv((long)Dd * CATd, 256), blk, 0, stream>>>(F(13), wt_dp,   (long)Dd * CATd);
    for (int i = 0; i < 3; ++i) {
        k_wcast<<<cdiv((long)Dd * Dd, 256), blk, 0, stream>>>(F(39 + 6 * i), wt_dwin[i],  (long)Dd * Dd);
        k_wcast<<<cdiv((long)Dd * Dd, 256), blk, 0, stream>>>(F(43 + 6 * i), wt_dwout[i], (long)Dd * Dd);
    }

    // ---- pipeline
    // x NCHW -> NHWC f16; x1 = x @ x_trans
    k_nchw2nhwc<<<cdiv((long)Ssz * INC, 256), blk, 0, stream>>>(F(0), xh);
    gemm(xh, INC, wt_xt, Ssz, Dd, INC, F(8), x1, nullptr, Dd, 0, nullptr, nullptr, false);

    // MSA: ln -> msa_s -> dense blocks -> dense_proj -> spatial attention
    k_layernorm<<<cdiv(Ssz, 8), blk, 0, stream>>>(x1, F(9), F(10), Ssz, Dd, a16, nullptr);
    gemm(a16, Dd, wt_msas, Ssz, Dd, Dd, F(12), nullptr, cat, CATd, 0, nullptr, nullptr, false);
    for (int i = 0; i < 3; ++i) {
        k_gelu16<<<cdiv((long)Ssz * Dd, 256), blk, 0, stream>>>(cat + (long)i * Dd, CATd, t16, Dd, Ssz, Dd);
        gemm(t16, Dd, wt_dwin[i], Ssz, Dd, Dd, F(40 + 6 * i), t1, nullptr, Dd, 0, nullptr, nullptr, false);
        k_dw3x3<0><<<cdiv((long)Ssz * Dd, 256), blk, 0, stream>>>(t1, Dd, F(41 + 6 * i), F(42 + 6 * i), t2h, Dd, 0);
        gemm(t2h, Dd, wt_dwout[i], Ssz, Dd, Dd, F(44 + 6 * i), nullptr, cat + (long)(i + 1) * Dd, CATd, 0, nullptr, nullptr, false);
    }
    gemm(cat, CATd, wt_dp, Ssz, Dd, CATd, F(14), zb, nullptr, Dd, 0, nullptr, nullptr, false);
    k_avgmax<<<cdiv(Ssz, 8), blk, 0, stream>>>(zb, am);
    k_spatatt<<<cdiv(Ssz, 256), blk, 0, stream>>>(am, F(15), att);
    k_msacomb<<<cdiv((long)Ssz * Dd, 256), blk, 0, stream>>>(zb, att, x1, F(36), x2, x2h);

    // Router
    gemm(x2h, Dd, wt_r1, Ssz, R1d, Dd, F(3), nullptr, r16, R1d, 1, nullptr, nullptr, false);
    k_router2<<<cdiv(Ssz, 256), blk, 0, stream>>>(r16, F(4), F(5), gates);

    // Q projection
    k_layernorm<<<cdiv(Ssz, 8), blk, 0, stream>>>(x2, F(16), F(17), Ssz, Dd, a16, nullptr);
    gemm(a16, Dd, wt_q, Ssz, Dd, Dd, F(19), nullptr, qh, Dd, 0, nullptr, nullptr, false);

    // Expert dictionary prep + fused cross-attention/soft-MoE combine
    k_layernorm<<<cdiv(KEXP * ENTd, 8), blk, 0, stream>>>(F(1), F(20), F(21), KEXP * ENTd, Dd, nullptr, ed);
    k_expert_kv<<<cdiv((long)KEXP * ENTd * Dd, 256), blk, 0, stream>>>(ed, F(22), F(23), kxh, vxT);
    {
        dim3 g(4096 / 128, Bsz * Ed);
        k_attention<<<g, blk, 0, stream>>>(qh, kxh, vxT, gates, F(6), ctxh);
    }

    // lin + residual(rs2)
    gemm(ctxh, Dd, wt_lin, Ssz, Dd, Dd, F(25), o2, nullptr, Dd, 0, x2, F(37), false);

    // ConvGELU MLP + residual(rs3)
    k_layernorm<<<cdiv(Ssz, 8), blk, 0, stream>>>(o2, F(26), F(27), Ssz, Dd, a16, nullptr);
    gemm(a16, Dd, wt_fc1, Ssz, CATd, Dd, F(29), hb, nullptr, CATd, 0, nullptr, nullptr, false);
    k_dw3x3<1><<<cdiv((long)Ssz * HIDd, 256), blk, 0, stream>>>(hb, CATd, F(30), F(31), mmid, HIDd, HIDd);
    gemm(mmid, HIDd, wt_fc2, Ssz, Dd, HIDd, F(33), o3, o3h, Dd, 0, o2, F(38), false);

    // Output projection, scattered straight into NCHW d_out
    gemm(o3h, Dd, wt_out, Ssz, OUTC, Dd, F(35), (float*)d_out, nullptr, 0, 0, nullptr, nullptr, true);
}
